// GNNEncoder_8315056685619
// MI455X (gfx1250) — matrix operand order repaired and checked by reading, not compiled, not run on hardware
//
#include <hip/hip_runtime.h>
#include <hip/hip_bf16.h>
#include <math.h>

#define NN   32768
#define BB   16
#define EE   262144
#define DDIM 128
#define SSUB 4
#define LLAY 3
#define HHID 128
#define DECO 512
#define PGRF 2048   // nodes per graph (N/B)

typedef __bf16 bf16_t;
typedef __attribute__((ext_vector_type(16))) __bf16 v16bf;
typedef __attribute__((ext_vector_type(8)))  __bf16 v8bf;
typedef __attribute__((ext_vector_type(8)))  float  v8f;

__device__ __forceinline__ float sigm(float x) { return 1.0f / (1.0f + __expf(-x)); }

// ---------------------------------------------------------------------------
// Pre-pack a 128 x Nreal weight matrix (optionally transposed source) into the
// WMMA B-fragment layout: [nTile][kbi][lane][16 elems], bf16.
//   mode 0: B[k][n] = W[k*Nreal + n]     (W is [128 x Nreal], row major)
//   mode 1: B[k][n] = W[n*128 + k]       (W is [Nreal x 128], use W^T)
// ---------------------------------------------------------------------------
__global__ void pack_b_kernel(const float* __restrict__ W, bf16_t* __restrict__ out,
                              int tilesN, int Nreal, int mode)
{
    int idx = blockIdx.x * blockDim.x + threadIdx.x;
    int total = tilesN * 4 * 32 * 16;
    if (idx >= total) return;
    int e     = idx & 15;
    int lane  = (idx >> 4) & 31;
    int kbi   = (idx >> 9) & 3;
    int nTile = idx >> 11;
    int n = nTile * 16 + (lane & 15);
    int k = kbi * 32 + (lane >> 4) * 16 + e;
    float v = 0.0f;
    if (n < Nreal) v = (mode == 0) ? W[k * Nreal + n] : W[n * 128 + k];
    out[idx] = (bf16_t)v;
}

// ---------------------------------------------------------------------------
// C[M x Nreal] (f32, stride Nreal) = A[M x 128] (bf16, row major) * Bpacked + bias
// One wave computes a 16 x 64 strip (4 WMMA tiles): the A fragment is loaded
// once per K-block and reused across 4 B tiles -> 16 WMMAs per wave, 4x less
// A traffic from L2.  NcPad must be a multiple of 64.
// ---------------------------------------------------------------------------
__global__ __launch_bounds__(256)
void gemm_bf16_wmma(const bf16_t* __restrict__ A, const bf16_t* __restrict__ Bp,
                    const float* __restrict__ bias, float* __restrict__ C,
                    int Mrows, int NcPad, int Nreal)
{
    const int lane = threadIdx.x & 31;
    const int wave = threadIdx.x >> 5;
    const int tilesN  = NcPad >> 4;
    const int groupsN = tilesN >> 2;                    // 4 tiles per group
    const int group = blockIdx.x * 8 + wave;
    const int totalGroups = (Mrows >> 4) * groupsN;
    if (group >= totalGroups) return;                   // wave-uniform: EXEC stays all-ones
    const int mTile = group / groupsN;
    const int nGrp  = group % groupsN;

    const int mLocal = lane & 15;
    const int kHalf  = lane >> 4;

    const bf16_t* aRow  = A  + (size_t)(mTile * 16 + mLocal) * 128;
    // base of the 4-tile B group for this lane
    const bf16_t* bGrp  = Bp + ((size_t)(nGrp * 4) * 4 * 32 + lane) * 16;

    v8f acc[4] = {{}, {}, {}, {}};
#pragma unroll
    for (int kb = 0; kb < 4; ++kb) {
        v8bf a0 = *(const v8bf*)(aRow + kb * 32 + kHalf * 16);
        v8bf a1 = *(const v8bf*)(aRow + kb * 32 + kHalf * 16 + 8);
        v16bf av;
#pragma unroll
        for (int i = 0; i < 8; ++i) { av[i] = a0[i]; av[i + 8] = a1[i]; }
#pragma unroll
        for (int t = 0; t < 4; ++t) {
            const bf16_t* bB = bGrp + ((size_t)t * 4 + kb) * 512;   // [tile][kbi] stride 32*16
            v8bf b0 = *(const v8bf*)(bB);
            v8bf b1 = *(const v8bf*)(bB + 8);
            v16bf bv;
#pragma unroll
            for (int i = 0; i < 8; ++i) { bv[i] = b0[i]; bv[i + 8] = b1[i]; }
            acc[t] = __builtin_amdgcn_wmma_f32_16x16x32_bf16(
                         false, av, false, bv, (short)0, acc[t], false, false);
        }
    }

#pragma unroll
    for (int t = 0; t < 4; ++t) {
        const int n = (nGrp * 4 + t) * 16 + (lane & 15);
        if (n < Nreal) {
            float bvv = bias ? bias[n] : 0.0f;
#pragma unroll
            for (int r = 0; r < 8; ++r) {
                int m = mTile * 16 + kHalf * 8 + r;
                C[(size_t)m * Nreal + n] = acc[t][r] + bvv;
            }
        }
    }
}

// ---------------------------------------------------------------------------
// Misc elementwise kernels
// ---------------------------------------------------------------------------
__global__ void embed_gather(const int* __restrict__ ids, const float* __restrict__ tab,
                             float* __restrict__ X, bf16_t* __restrict__ XB)
{
    int idx = blockIdx.x * blockDim.x + threadIdx.x;
    if (idx >= NN * DDIM) return;
    int node = idx >> 7, d = idx & 127;
    float v = tab[(size_t)ids[node] * DDIM + d];
    X[idx] = v; XB[idx] = (bf16_t)v;
}

__global__ void edge_weight_kernel(const int* __restrict__ attr, const float* __restrict__ tab,
                                   float* __restrict__ ew)
{
    int e = blockIdx.x * blockDim.x + threadIdx.x;
    if (e >= EE) return;
    ew[e] = tab[attr[e]];
}

__global__ void copy_init_h(const float* __restrict__ X, const bf16_t* __restrict__ XB,
                            float* __restrict__ H, bf16_t* __restrict__ HB)
{
    int idx = blockIdx.x * blockDim.x + threadIdx.x;
    if (idx >= NN * DDIM) return;
    H[idx] = X[idx]; HB[idx] = XB[idx];
}

__global__ void zero_f32(float* __restrict__ p, int n)
{
    int idx = blockIdx.x * blockDim.x + threadIdx.x;
    if (idx < n) p[idx] = 0.0f;
}

__global__ void cvt_bf16(const float* __restrict__ src, bf16_t* __restrict__ dst, int n)
{
    int idx = blockIdx.x * blockDim.x + threadIdx.x;
    if (idx < n) dst[idx] = (bf16_t)src[idx];
}

// msg = m[src]*w ; agg[tgt] += msg   (f32 atomics, 4 elems per thread)
__global__ void edge_scatter(const float* __restrict__ Msg,
                             const int* __restrict__ src, const int* __restrict__ tgt,
                             const float* __restrict__ ew, const int* __restrict__ esub,
                             int sub, float* __restrict__ agg)
{
    int idx = blockIdx.x * blockDim.x + threadIdx.x;   // EE * 32 threads
    int e = idx >> 5;
    if (e >= EE) return;
    if (sub >= 0 && esub[e] != sub) return;
    float w = ew[e];
    if (w == 0.0f) return;
    int c = (idx & 31) * 4;
    const float4 v = *(const float4*)(Msg + (size_t)src[e] * DDIM + c);
    float* dst = agg + (size_t)tgt[e] * DDIM + c;
    __hip_atomic_fetch_add(dst + 0, v.x * w, __ATOMIC_RELAXED, __HIP_MEMORY_SCOPE_AGENT);
    __hip_atomic_fetch_add(dst + 1, v.y * w, __ATOMIC_RELAXED, __HIP_MEMORY_SCOPE_AGENT);
    __hip_atomic_fetch_add(dst + 2, v.z * w, __ATOMIC_RELAXED, __HIP_MEMORY_SCOPE_AGENT);
    __hip_atomic_fetch_add(dst + 3, v.w * w, __ATOMIC_RELAXED, __HIP_MEMORY_SCOPE_AGENT);
}

// GRU update, in-place on h; also refresh bf16 shadow
__global__ void gru_update(const float* __restrict__ gi, const float* __restrict__ gh,
                           float* __restrict__ H, bf16_t* __restrict__ HB)
{
    int idx = blockIdx.x * blockDim.x + threadIdx.x;
    if (idx >= NN * DDIM) return;
    int node = idx >> 7, d = idx & 127;
    size_t b3 = (size_t)node * 384;
    float ir = gi[b3 + d], iz = gi[b3 + 128 + d], in_ = gi[b3 + 256 + d];
    float hr = gh[b3 + d], hz = gh[b3 + 128 + d], hn  = gh[b3 + 256 + d];
    float r = sigm(ir + hr);
    float z = sigm(iz + hz);
    float n = tanhf(in_ + r * hn);
    float h = H[idx];
    float o = (1.0f - z) * n + z * h;
    H[idx] = o; HB[idx] = (bf16_t)o;
}

__global__ void sigmoid_inplace(float* __restrict__ p, int n)
{
    int idx = blockIdx.x * blockDim.x + threadIdx.x;
    if (idx < n) p[idx] = sigm(p[idx]);
}

// g[node] = sigmoid( dot(U[node,:300], w2) + b2 )
__global__ void gate_dot(const float* __restrict__ U, const float* __restrict__ w2,
                         const float* __restrict__ b2, float* __restrict__ gv)
{
    int node = blockIdx.x * blockDim.x + threadIdx.x;
    if (node >= NN) return;
    const float* row = U + (size_t)node * 300;
    float acc = 0.0f;
    for (int k = 0; k < 300; ++k) acc += row[k] * w2[k];
    gv[node] = sigm(acc + b2[0]);
}

__global__ void seg_max(const float* __restrict__ gv, float* __restrict__ gmax)
{
    __shared__ float red[256];
    int b = blockIdx.x;
    float m = -1e30f;
    for (int i = threadIdx.x; i < PGRF; i += 256) m = fmaxf(m, gv[b * PGRF + i]);
    red[threadIdx.x] = m; __syncthreads();
    for (int s = 128; s > 0; s >>= 1) {
        if ((int)threadIdx.x < s) red[threadIdx.x] = fmaxf(red[threadIdx.x], red[threadIdx.x + s]);
        __syncthreads();
    }
    if (threadIdx.x == 0) gmax[b] = red[0];
}

__global__ void seg_expsum(const float* __restrict__ gv, const float* __restrict__ gmax,
                           float* __restrict__ ev, float* __restrict__ den)
{
    __shared__ float red[256];
    int b = blockIdx.x;
    float gm = gmax[b];
    float s = 0.0f;
    for (int i = threadIdx.x; i < PGRF; i += 256) {
        float e = __expf(gv[b * PGRF + i] - gm);
        ev[b * PGRF + i] = e;
        s += e;
    }
    red[threadIdx.x] = s; __syncthreads();
    for (int st = 128; st > 0; st >>= 1) {
        if ((int)threadIdx.x < st) red[threadIdx.x] += red[threadIdx.x + st];
        __syncthreads();
    }
    if (threadIdx.x == 0) den[b] = red[0];
}

// pool[b][d] = (1/den[b]) * sum_n ev[n] * H[n][d]   (nodes of graph b contiguous)
__global__ void attn_pool(const float* __restrict__ ev, const float* __restrict__ den,
                          const float* __restrict__ H, float* __restrict__ outp)
{
    int b = blockIdx.x, d = threadIdx.x;   // 128 threads
    float acc = 0.0f;
    for (int n = 0; n < PGRF; ++n)
        acc += ev[b * PGRF + n] * H[(size_t)(b * PGRF + n) * DDIM + d];
    outp[b * DDIM + d] = acc / den[b];
}

// tiny 4-step LSTM over seq[4][16][128]; writes last hidden state
__global__ void lstm_seq(const float* __restrict__ seq,
                         const float* __restrict__ wih, const float* __restrict__ whh,
                         const float* __restrict__ bih, const float* __restrict__ bhh,
                         float* __restrict__ last)
{
    __shared__ float sh[HHID], sc[HHID], sx[HHID];
    int b = blockIdx.x, j = threadIdx.x;   // 128 threads
    sh[j] = 0.0f; sc[j] = 0.0f;
    __syncthreads();
    for (int t = 0; t < SSUB; ++t) {
        sx[j] = seq[((size_t)t * BB + b) * DDIM + j];
        __syncthreads();
        float gi = bih[j]       + bhh[j];
        float gf = bih[128 + j] + bhh[128 + j];
        float gg = bih[256 + j] + bhh[256 + j];
        float go = bih[384 + j] + bhh[384 + j];
        for (int k = 0; k < 128; ++k) {
            float xv = sx[k], hv = sh[k];
            gi += xv * wih[(0 * 128 + j) * 128 + k] + hv * whh[(0 * 128 + j) * 128 + k];
            gf += xv * wih[(1 * 128 + j) * 128 + k] + hv * whh[(1 * 128 + j) * 128 + k];
            gg += xv * wih[(2 * 128 + j) * 128 + k] + hv * whh[(2 * 128 + j) * 128 + k];
            go += xv * wih[(3 * 128 + j) * 128 + k] + hv * whh[(3 * 128 + j) * 128 + k];
        }
        __syncthreads();
        float c = sigm(gf) * sc[j] + sigm(gi) * tanhf(gg);
        float h = sigm(go) * tanhf(c);
        sc[j] = c; sh[j] = h;
        __syncthreads();
    }
    last[b * HHID + j] = sh[j];
}

// out[b][o] = concat(last, gpool)[b] @ fc_w + fc_b
__global__ void fc_out(const float* __restrict__ last, const float* __restrict__ gpool,
                       const float* __restrict__ w, const float* __restrict__ bias,
                       float* __restrict__ out)
{
    int idx = blockIdx.x * blockDim.x + threadIdx.x;
    if (idx >= BB * DECO) return;
    int b = idx >> 9, o2 = idx & 511;
    float acc = bias[o2];
    for (int k = 0; k < 128; ++k) acc += last[b * 128 + k]  * w[k * DECO + o2];
    for (int k = 0; k < 128; ++k) acc += gpool[b * 128 + k] * w[(128 + k) * DECO + o2];
    out[idx] = acc;
}

// ---------------------------------------------------------------------------
extern "C" void kernel_launch(void* const* d_in, const int* in_sizes, int n_in,
                              void* d_out, int out_size, void* d_ws, size_t ws_size,
                              hipStream_t stream)
{
    (void)in_sizes; (void)n_in; (void)out_size; (void)ws_size;

    const int*   x_ids  = (const int*)d_in[0];
    const int*   eidx   = (const int*)d_in[1];           // [2,E]: src then tgt
    const int*   eattr  = (const int*)d_in[2];
    const int*   esub   = (const int*)d_in[3];
    const float* embed  = (const float*)d_in[7];
    const float* eembed = (const float*)d_in[8];
    const float* ggnn   = (const float*)d_in[9];
    const float* w_ih   = (const float*)d_in[10];
    const float* w_hh   = (const float*)d_in[11];
    const float* b_ih   = (const float*)d_in[12];
    const float* b_hh   = (const float*)d_in[13];
    const float* gw1    = (const float*)d_in[14];
    const float* gb1    = (const float*)d_in[15];
    const float* gw2    = (const float*)d_in[16];
    const float* gb2    = (const float*)d_in[17];
    const float* lwih   = (const float*)d_in[18];
    const float* lwhh   = (const float*)d_in[19];
    const float* lbih   = (const float*)d_in[20];
    const float* lbhh   = (const float*)d_in[21];
    const float* fcw    = (const float*)d_in[22];
    const float* fcb    = (const float*)d_in[23];
    const int* src = eidx;
    const int* tgt = eidx + EE;

    // ---- workspace carve-out ------------------------------------------------
    char* ws = (char*)d_ws;
    size_t off = 0;
    auto carve = [&](size_t bytes) -> char* {
        char* p = ws + off;
        off += (bytes + 255) & ~(size_t)255;
        return p;
    };
    float*  X     = (float*)carve((size_t)NN * DDIM * 4);
    bf16_t* XB    = (bf16_t*)carve((size_t)NN * DDIM * 2);
    float*  H     = (float*)carve((size_t)NN * DDIM * 4);
    bf16_t* HB    = (bf16_t*)carve((size_t)NN * DDIM * 2);
    float*  M     = (float*)carve((size_t)NN * DDIM * 4);
    float*  AGG   = (float*)carve((size_t)NN * DDIM * 4);
    bf16_t* AGGB  = (bf16_t*)carve((size_t)NN * DDIM * 2);
    float*  GH    = (float*)carve((size_t)NN * 384 * 4);
    float*  GI    = (float*)carve((size_t)NN * 384 * 4);   // reused as T (N x 300) in attention
    float*  EW    = (float*)carve((size_t)EE * 4);
    float*  GV    = (float*)carve((size_t)NN * 4);
    float*  EV    = (float*)carve((size_t)NN * 4);
    float*  GMAX  = (float*)carve(BB * 4);
    float*  DEN   = (float*)carve(BB * 4);
    float*  SEQ   = (float*)carve((size_t)SSUB * BB * DDIM * 4);
    float*  GPOOL = (float*)carve((size_t)BB * DDIM * 4);
    float*  LAST  = (float*)carve((size_t)BB * HHID * 4);
    bf16_t* PKG   = (bf16_t*)carve((size_t)LLAY * 8 * 4 * 512 * 2);   // 3 x [8 tiles]
    bf16_t* PKIH  = (bf16_t*)carve((size_t)24 * 4 * 512 * 2);
    bf16_t* PKHH  = (bf16_t*)carve((size_t)24 * 4 * 512 * 2);
    bf16_t* PKW1  = (bf16_t*)carve((size_t)20 * 4 * 512 * 2);         // padded to 320 cols

    const int TPB = 256;
    auto nb = [](long n, int t) { return (int)((n + t - 1) / t); };

    // ---- weight packing (launch-invariant inputs, deterministic) -----------
    for (int l = 0; l < LLAY; ++l)
        pack_b_kernel<<<nb(8 * 4 * 512, TPB), TPB, 0, stream>>>(
            ggnn + (size_t)l * 128 * 128, PKG + (size_t)l * 8 * 4 * 512, 8, 128, 0);
    pack_b_kernel<<<nb(24 * 4 * 512, TPB), TPB, 0, stream>>>(w_ih, PKIH, 24, 384, 1);
    pack_b_kernel<<<nb(24 * 4 * 512, TPB), TPB, 0, stream>>>(w_hh, PKHH, 24, 384, 1);
    pack_b_kernel<<<nb(20 * 4 * 512, TPB), TPB, 0, stream>>>(gw1, PKW1, 20, 300, 0);

    // ---- inputs -------------------------------------------------------------
    embed_gather<<<nb((long)NN * DDIM, TPB), TPB, 0, stream>>>(x_ids, embed, X, XB);
    edge_weight_kernel<<<nb(EE, TPB), TPB, 0, stream>>>(eattr, eembed, EW);

    const long ND = (long)NN * DDIM;
    // blocks = ceil(mTiles * groupsN / 8 waves)
    const int gemm128 = nb((long)(NN / 16) * (128 / 64), 8);
    const int gemm384 = nb((long)(NN / 16) * (384 / 64), 8);
    const int gemm320 = nb((long)(NN / 16) * (320 / 64), 8);

    // ---- 5 conv_pool passes: s = 0..3 masked, s = -1 full -------------------
    for (int cfg = 0; cfg < SSUB + 1; ++cfg) {
        int sub = (cfg < SSUB) ? cfg : -1;
        float* pool_out = (cfg < SSUB) ? (SEQ + (size_t)cfg * BB * DDIM) : GPOOL;

        copy_init_h<<<nb(ND, TPB), TPB, 0, stream>>>(X, XB, H, HB);

        for (int l = 0; l < LLAY; ++l) {
            // m = h @ W[l]
            gemm_bf16_wmma<<<gemm128, 256, 0, stream>>>(
                HB, PKG + (size_t)l * 8 * 4 * 512, nullptr, M, NN, 128, 128);
            // gh = h @ w_hh^T + b_hh
            gemm_bf16_wmma<<<gemm384, 256, 0, stream>>>(
                HB, PKHH, b_hh, GH, NN, 384, 384);
            // agg = scatter_add(m[src] * ew_masked, tgt)
            zero_f32<<<nb(ND, TPB), TPB, 0, stream>>>(AGG, (int)ND);
            edge_scatter<<<nb((long)EE * 32, TPB), TPB, 0, stream>>>(
                M, src, tgt, EW, esub, sub, AGG);
            cvt_bf16<<<nb(ND, TPB), TPB, 0, stream>>>(AGG, AGGB, (int)ND);
            // gi = agg @ w_ih^T + b_ih
            gemm_bf16_wmma<<<gemm384, 256, 0, stream>>>(
                AGGB, PKIH, b_ih, GI, NN, 384, 384);
            // h = GRU(gi, gh, h)
            gru_update<<<nb(ND, TPB), TPB, 0, stream>>>(GI, GH, H, HB);
        }

        // global attention pooling
        float* T = GI;   // reuse: N x 300, stride 300
        gemm_bf16_wmma<<<gemm320, 256, 0, stream>>>(HB, PKW1, gb1, T, NN, 320, 300);
        sigmoid_inplace<<<nb((long)NN * 300, TPB), TPB, 0, stream>>>(T, NN * 300);
        gate_dot<<<nb(NN, TPB), TPB, 0, stream>>>(T, gw2, gb2, GV);
        seg_max<<<BB, 256, 0, stream>>>(GV, GMAX);
        seg_expsum<<<BB, 256, 0, stream>>>(GV, GMAX, EV, DEN);
        attn_pool<<<BB, DDIM, 0, stream>>>(EV, DEN, H, pool_out);
    }

    // ---- LSTM over seq + final FC ------------------------------------------
    lstm_seq<<<BB, HHID, 0, stream>>>(SEQ, lwih, lwhh, lbih, lbhh, LAST);
    fc_out<<<nb((long)BB * DECO, TPB), TPB, 0, stream>>>(LAST, GPOOL, fcw, fcb, (float*)d_out);
}
